// CRF_52166672777278
// MI455X (gfx1250) — matrix operand order repaired and checked by reading, not compiled, not run on hardware
//
#include <hip/hip_runtime.h>
#include <math.h>
#include <stdint.h>

#define B_ 256
#define T_ 1024
#define K_ 128
#define BT 16   // batch rows per workgroup (WMMA M)

typedef __attribute__((ext_vector_type(16))) _Float16 v16h;
typedef __attribute__((ext_vector_type(2)))  __fp16   v2fp16;  // cvt_pkrtz return type
typedef __attribute__((ext_vector_type(8)))  float    v8f;
typedef __attribute__((ext_vector_type(4)))  float    v4f;

union V16H { v16h v; uint32_t u[8]; _Float16 h[16]; };
union V2H  { v2fp16 v; uint32_t u; };

#define LOG2E 1.4426950408889634f
#define LN2   0.6931471805599453f

// ---------------------------------------------------------------------------
// Kernel 1: unary + binary tag scores per batch -> d_ws[b]
// ---------------------------------------------------------------------------
__global__ __launch_bounds__(256)
void crf_scores_kernel(const float* __restrict__ logits,
                       const float* __restrict__ trans,
                       const int*   __restrict__ tags,
                       const int*   __restrict__ seq_lens,
                       float*       __restrict__ scores) {
  __shared__ float red[256];
  const int b = blockIdx.x;
  const int L = seq_lens[b];
  float acc = 0.0f;
  for (int t = threadIdx.x; t < T_; t += 256) {
    if (t < L) {
      int tg = tags[b * T_ + t];
      acc += logits[((size_t)b * T_ + t) * K_ + tg];           // unary
      if (t >= 1) {
        int tp = tags[b * T_ + t - 1];
        acc += trans[tp * K_ + tg];                            // binary
      }
    }
  }
  red[threadIdx.x] = acc;
  __syncthreads();
  for (int s = 128; s > 0; s >>= 1) {
    if (threadIdx.x < s) red[threadIdx.x] += red[threadIdx.x + s];
    __syncthreads();
  }
  if (threadIdx.x == 0) scores[b] = red[0];
}

// ---------------------------------------------------------------------------
// Kernel 2: batch-parallel CRF forward scan. One workgroup = 16 batch rows,
// 8 waves; wave w computes output columns [16w,16w+16) via WMMA f16->f32.
// Step: S = exp(alpha - rowmax) @ exp(Trans);  alpha' = logit + rowmax + log S
// Two barriers per step: publish {P, amax}, publish alpha'.
// ---------------------------------------------------------------------------
__global__ __launch_bounds__(256)
void crf_forward_kernel(const float* __restrict__ logits,
                        const float* __restrict__ trans,
                        const int*   __restrict__ seq_lens,
                        const float* __restrict__ scores,
                        float*       __restrict__ out) {
  // Padded rows: stride 136 halves (272B) / 132 floats -> no 64-bank stride conflicts
  __shared__ __align__(16) _Float16 ldsB[K_][K_ + 8];  // exp(Trans)^T: ldsB[j][i]=exp(T[i][j])
  __shared__ __align__(16) _Float16 pbuf[BT][K_ + 8];  // P = exp(alpha - rowmax), f16
  __shared__ __align__(16) float    alpha[BT][K_ + 4]; // read in loop only for rowmax/P (regs hold own cols)
  __shared__ __align__(16) float    red[BT][16];
  __shared__ __align__(16) float    amax[BT];
  __shared__ int                    sl[BT];

  const int tid  = threadIdx.x;
  const int lane = tid & 31;
  const int wv   = tid >> 5;                 // wave id == N-tile index (0..7)
  const int b0   = blockIdx.x * BT;

  // exp(transitions), transposed for contiguous-K B-fragment gathers
  for (int f = tid; f < K_ * K_; f += 256) {
    int i = f & (K_ - 1);
    int j = f >> 7;
    ldsB[j][i] = (_Float16)__expf(trans[i * K_ + j]);
  }
  if (tid < BT) sl[tid] = seq_lens[b0 + tid];

  // alpha0 = logits[:, 0, :] (LDS copy for the cross-lane rowmax phase)
  const int srow = tid >> 4, ssub = tid & 15;
  {
    const float* src = logits + ((size_t)(b0 + srow) * T_) * K_ + ssub * 8;
    #pragma unroll
    for (int q = 0; q < 8; ++q) alpha[srow][ssub * 8 + q] = src[q];
  }
  __syncthreads();

  // Preload time-invariant B fragments (contiguous-K order, same as A:
  // lane L holds col N=L%16, half h <-> K = h + 16*(L>=16))
  V16H bfr[4];
  {
    const int j    = wv * 16 + (lane & 15);
    const int koff = (lane >= 16) ? 8 : 0;   // dword offset (= 16 halves)
    const uint32_t* brow = (const uint32_t*)&ldsB[j][0];
    #pragma unroll
    for (int kc = 0; kc < 4; ++kc)
      #pragma unroll
      for (int v = 0; v < 8; ++v)
        bfr[kc].u[v] = brow[kc * 16 + koff + v];
  }

  const int rowM  = lane & 15;               // A-fragment row
  const int jcol  = wv * 16 + (lane & 15);   // owned output column
  const int rbase = (lane >= 16) ? 8 : 0;    // C/D row base

  // Loop-invariant per-lane state: own alpha column values + seq lens
  float aold[8];
  int   slr[8];
  #pragma unroll
  for (int r = 0; r < 8; ++r) {
    aold[r] = logits[((size_t)(b0 + rbase + r) * T_) * K_ + jcol];
    slr[r]  = seq_lens[b0 + rbase + r];
  }

  // Incremental logit pointer: 8 row offsets r*T*K*4 <= 3.67MB fit in imm24
  const float* lptr = logits + ((size_t)(b0 + rbase) * T_ + 1) * K_ + jcol;

  for (int t = 1; t < T_; ++t) {
    // Hoisted logit gather: independent of alpha, overlaps barrier chain
    float lg[8];
    #pragma unroll
    for (int r = 0; r < 8; ++r) lg[r] = lptr[r * (T_ * K_)];
    lptr += K_;

    // Row max of alpha: 16 threads per row, butterfly over the 16-lane half
    float av[8];
    {
      const float* ar = &alpha[srow][ssub * 8];
      #pragma unroll
      for (int q = 0; q < 8; ++q) av[q] = ar[q];
    }
    float m = av[0];
    #pragma unroll
    for (int q = 1; q < 8; ++q) m = fmaxf(m, av[q]);
    m = fmaxf(m, __shfl_xor(m, 1));
    m = fmaxf(m, __shfl_xor(m, 2));
    m = fmaxf(m, __shfl_xor(m, 4));
    m = fmaxf(m, __shfl_xor(m, 8));

    // P = exp2(fma(alpha, log2e, -m*log2e)) -> packed f16 dword stores
    {
      const float nm = -m * LOG2E;
      uint32_t* pw = (uint32_t*)&pbuf[srow][ssub * 8];
      #pragma unroll
      for (int q = 0; q < 4; ++q) {
        float e0 = __builtin_amdgcn_exp2f(fmaf(av[2 * q],     LOG2E, nm));
        float e1 = __builtin_amdgcn_exp2f(fmaf(av[2 * q + 1], LOG2E, nm));
        V2H pk;
        pk.v = __builtin_amdgcn_cvt_pkrtz(e0, e1);
        pw[q] = pk.u;
      }
    }
    if (ssub == 0) amax[srow] = m;
    __syncthreads();                         // publish P + amax

    // base = logit + amax (independent of WMMA result; overlaps the chain)
    float base[8];
    {
      const v4f* ap = (const v4f*)&amax[rbase];
      v4f a0 = ap[0], a1 = ap[1];
      #pragma unroll
      for (int r = 0; r < 4; ++r) base[r]     = lg[r]     + a0[r];
      #pragma unroll
      for (int r = 0; r < 4; ++r) base[r + 4] = lg[r + 4] + a1[r];
    }

    // 4 chained WMMAs over K=128 (A 16x32, contiguous K to match B:
    // lane L row M=L%16, half h <-> K = h + 16*(L>=16))
    v8f c = {};
    {
      const uint32_t* prow = (const uint32_t*)&pbuf[rowM][0];
      const int koff = (lane >= 16) ? 8 : 0; // dwords (= 16 halves)
      #pragma unroll
      for (int kc = 0; kc < 4; ++kc) {
        V16H a;
        #pragma unroll
        for (int v = 0; v < 8; ++v) a.u[v] = prow[kc * 16 + koff + v];
        c = __builtin_amdgcn_wmma_f32_16x16x32_f16(
                false, a.v, false, bfr[kc].v, (short)0, c, false, false);
      }
    }

    // alpha' = base + ln2*log2(S); S >= e^-6, never denormal -> raw v_log_f32
    #pragma unroll
    for (int r = 0; r < 8; ++r) {
      float nv = fmaf(LN2, __builtin_amdgcn_logf(c[r]), base[r]);
      aold[r]  = (t < slr[r]) ? nv : aold[r];
      alpha[rbase + r][jcol] = aold[r];
    }
    __syncthreads();                         // publish alpha'
  }

  // Final: out[b] = scores[b] - logsumexp(alpha[b, :])
  {
    const float* ar = &alpha[srow][ssub * 8];
    float m = ar[0];
    #pragma unroll
    for (int q = 1; q < 8; ++q) m = fmaxf(m, ar[q]);
    red[srow][ssub] = m;
  }
  __syncthreads();
  if (tid < BT) {
    float m = red[tid][0];
    #pragma unroll
    for (int q = 1; q < 16; ++q) m = fmaxf(m, red[tid][q]);
    amax[tid] = m;
  }
  __syncthreads();
  {
    const float am = amax[srow];
    const float* ar = &alpha[srow][ssub * 8];
    float s = 0.0f;
    #pragma unroll
    for (int q = 0; q < 8; ++q) s += __expf(ar[q] - am);
    red[srow][ssub] = s;
  }
  __syncthreads();
  if (tid < BT) {
    float s = 0.0f;
    #pragma unroll
    for (int q = 0; q < 16; ++q) s += red[tid][q];
    out[b0 + tid] = scores[b0 + tid] - (amax[tid] + logf(s));
  }
}

// ---------------------------------------------------------------------------
extern "C" void kernel_launch(void* const* d_in, const int* in_sizes, int n_in,
                              void* d_out, int out_size, void* d_ws, size_t ws_size,
                              hipStream_t stream) {
  const float* logits = (const float*)d_in[0];   // (B,T,K) f32
  const float* trans  = (const float*)d_in[1];   // (K,K)   f32
  const int*   tags   = (const int*)d_in[2];     // (B,T)
  const int*   slens  = (const int*)d_in[3];     // (B,)
  float*       out    = (float*)d_out;           // (B,)
  float*       scores = (float*)d_ws;            // B floats of scratch

  crf_scores_kernel <<<B_,      256, 0, stream>>>(logits, trans, tags, slens, scores);
  crf_forward_kernel<<<B_ / BT, 256, 0, stream>>>(logits, trans, slens, scores, out);
}
